// Gemma3nAudioAttention_704374636653
// MI455X (gfx1250) — matrix operand order repaired and checked by reading, not compiled, not run on hardware
//
#include <hip/hip_runtime.h>
#include <math.h>

typedef _Float16 f16;
typedef _Float16 v16h __attribute__((ext_vector_type(16)));
typedef float    v8f  __attribute__((ext_vector_type(8)));
typedef unsigned int v4u __attribute__((ext_vector_type(4)));
typedef int      v4i __attribute__((ext_vector_type(4)));
typedef int      v8i __attribute__((ext_vector_type(8)));

#define HID    1536
#define HEADS  8
#define DH     192
#define CHUNK  12
#define PASTW  12
#define CTXW   24
#define KROW   204    // padded LDS row stride (halves) produced by TDM padding

union FragH { v16h v; unsigned int u[8]; f16 h[16]; };
union U32H  { unsigned int w; f16 h[2]; };

// ---------------------------------------------------------------------------
// TDM: async 2-D f16 tile load global -> LDS (CDNA5 Tensor Data Mover).
// Descriptor per ISA ch.8: group0 = {count, lds_addr, global_addr, type=2},
// group1 = {data_size=2B, pad_enable, pad_interval=16 DW, pad_amount=1 DW,
//           tensor_dim0/1 = tile dims (no OOB), tile_dim0/1, dim0 stride}.
// Padding inserts 1 DWORD after every 64 data bytes -> LDS offset of element
// (row, d) = KROW*row + d + 2*(d>>5)  (halves), bank-conflict-free frag reads.
// This toolchain's builtin is the 6-arg form:
//   (uint32x4 g0, int32x8 g1, int32x4, int32x4, int32x8, i32 cpol)
// ---------------------------------------------------------------------------
#if defined(__has_builtin)
#if __has_builtin(__builtin_amdgcn_tensor_load_to_lds) && \
    __has_builtin(__builtin_amdgcn_s_wait_tensorcnt)
#define USE_TDM 1
#endif
#endif
#ifndef USE_TDM
#define USE_TDM 0
#endif

#if USE_TDM
__device__ __forceinline__ void tdm_load_2d_f16(const f16* gsrc,
                                                unsigned lds_byte_off,
                                                unsigned rows,
                                                unsigned row_stride_elems) {
  unsigned long ga = (unsigned long)gsrc;
  v4u g0;
  g0[0] = 1u;                                        // count=1 (valid D#)
  g0[1] = lds_byte_off;                              // lds_addr
  g0[2] = (unsigned)(ga & 0xffffffffu);              // global_addr[31:0]
  g0[3] = (unsigned)((ga >> 32) & 0x01ffffffu)       // global_addr[56:32]
          | 0x80000000u;                             // type=2 ("image")
  v8i g1;
  g1[0] = (int)((1u << 16)      /* data_size = 2 bytes  */
              | (1u << 20)      /* pad_enable           */
              | (3u << 22));    /* pad_interval = 16 DW */ /* pad_amount=1DW */
  g1[1] = (int)((DH & 0xffffu) << 16);               // tensor_dim0 lo16
  g1[2] = (int)(((DH >> 16) & 0xffffu) | ((rows & 0xffffu) << 16)); // dim0 hi | dim1 lo
  g1[3] = (int)(((rows >> 16) & 0xffffu) | ((DH & 0xffffu) << 16)); // dim1 hi | tile_dim0
  g1[4] = (int)(rows & 0xffffu);                     // tile_dim1 | tile_dim2=0
  g1[5] = (int)row_stride_elems;                     // tensor_dim0_stride lo32
  g1[6] = 0;                                         // stride hi | dim1_stride lo
  g1[7] = 0;
  v4i z4 = {0, 0, 0, 0};
  v8i z8 = {0, 0, 0, 0, 0, 0, 0, 0};
  __builtin_amdgcn_tensor_load_to_lds(g0, g1, z4, z4, z8, 0);
}
#endif

// ---------------------------------------------------------------------------
// dscale[d] = D^-0.5 * (1/ln2) * softplus(per_dim_scale[d])
// ---------------------------------------------------------------------------
__global__ void gemma3n_dscale_kernel(const float* __restrict__ pds,
                                      float* __restrict__ dsc) {
  int d = threadIdx.x;
  if (d < DH) {
    float sp = log1pf(__expf(pds[d]));
    dsc[d] = 0.07216878364870323f * 1.4426950408889634f * sp;
  }
}

// ---------------------------------------------------------------------------
// sin_emb[f, n] = timing_signal(pos[f]) @ pos_w, stored f16. One block per f.
// ---------------------------------------------------------------------------
__global__ __launch_bounds__(256) void gemma3n_posemb_kernel(
    const float* __restrict__ pos_w, f16* __restrict__ semb) {
  __shared__ float tim[HID];
  const int f = blockIdx.x;                 // 0..12
  const float p = (float)(PASTW - f);       // pos = 12,11,...,0
  const float inc = logf(10000.0f) / 767.0f;
  for (int j = threadIdx.x; j < HID / 2; j += blockDim.x) {
    float st = p * __expf(-(float)j * inc);
    tim[j]           = __sinf(st);
    tim[j + HID / 2] = __cosf(st);
  }
  __syncthreads();
  for (int n = threadIdx.x; n < HID; n += blockDim.x) {
    float acc = 0.f;
    for (int k = 0; k < HID; ++k)
      acc = fmaf(tim[k], pos_w[(size_t)k * HID + n], acc);
    semb[(size_t)f * HID + n] = (f16)acc;
  }
}

// ---------------------------------------------------------------------------
// Fused QKV projection: [Mrows x 1536] @ [1536 x 1536] x3, f32 in, f16 out.
// 128x128 tile, 8 waves, wave = 32x64 (2x4 WMMA frags), K=32/step,
// software-pipelined: next K-tile global loads overlap current WMMAs.
// ---------------------------------------------------------------------------
__global__ __launch_bounds__(256) void gemma3n_qkv_gemm_kernel(
    const float* __restrict__ X,
    const float* __restrict__ Wq, const float* __restrict__ Wk,
    const float* __restrict__ Wv, const float* __restrict__ dsc,
    f16* __restrict__ Q, f16* __restrict__ Kd, f16* __restrict__ Vd,
    int Mrows) {
  __shared__ f16 As[128][36];   // [m][k], stride 36 halves
  __shared__ f16 Bs[128][36];   // [n][k] (transposed weight tile)

  const int tid  = threadIdx.x;
  const int lane = tid & 31;
  const int wv   = tid >> 5;
  const int wvm  = wv & 3;       // 4 waves along M
  const int wvn  = wv >> 2;      // 2 waves along N
  const int half = lane >> 4;
  const int l16  = lane & 15;

  const int mat   = blockIdx.x / 12;            // 0=q 1=k 2=v
  const int nco   = (blockIdx.x % 12) * 128;
  const int mBase = blockIdx.y * 128;

  const float* W = (mat == 0) ? Wq : (mat == 1) ? Wk : Wv;
  f16* Out       = (mat == 0) ? Q  : (mat == 1) ? Kd : Vd;

  v8f acc[2][4] = {};
  float4 aR[4], bR[4];

  // prologue: prefetch K-tile 0
  #pragma unroll
  for (int p = 0; p < 4; ++p) {
    int idx = tid + p * 256;
    int row = idx >> 3, c4 = idx & 7;
    int m = mBase + row; int mc = (m < Mrows) ? m : (Mrows - 1);
    aR[p] = *(const float4*)(X + (size_t)mc * HID + c4 * 4);
    int kr = idx >> 5, c4b = idx & 31;
    bR[p] = *(const float4*)(W + (size_t)kr * HID + nco + c4b * 4);
  }

  for (int kt = 0; kt < HID / 32; ++kt) {
    // commit staged registers to LDS (f32 -> f16)
    #pragma unroll
    for (int p = 0; p < 4; ++p) {
      int idx = tid + p * 256;
      int row = idx >> 3, c4 = idx & 7;
      f16* dst = &As[row][c4 * 4];
      dst[0] = (f16)aR[p].x; dst[1] = (f16)aR[p].y;
      dst[2] = (f16)aR[p].z; dst[3] = (f16)aR[p].w;
      int kr = idx >> 5, c4b = idx & 31, n = c4b * 4;
      Bs[n + 0][kr] = (f16)bR[p].x; Bs[n + 1][kr] = (f16)bR[p].y;
      Bs[n + 2][kr] = (f16)bR[p].z; Bs[n + 3][kr] = (f16)bR[p].w;
    }
    __syncthreads();

    // prefetch next K-tile while WMMAs run
    int k0n = (kt + 1) * 32;
    if (k0n < HID) {
      #pragma unroll
      for (int p = 0; p < 4; ++p) {
        int idx = tid + p * 256;
        int row = idx >> 3, c4 = idx & 7;
        int m = mBase + row; int mc = (m < Mrows) ? m : (Mrows - 1);
        aR[p] = *(const float4*)(X + (size_t)mc * HID + k0n + c4 * 4);
        int kr = idx >> 5, c4b = idx & 31;
        bR[p] = *(const float4*)(W + (size_t)(k0n + kr) * HID + nco + c4b * 4);
      }
    }

    FragH a[2], b[4];
    #pragma unroll
    for (int i = 0; i < 2; ++i) {
      const f16* base = &As[32 * wvm + 16 * i + l16][0];
      #pragma unroll
      for (int v = 0; v < 8; ++v) {       // A: K=2v (+16 upper half)
        int kb = 16 * half + 2 * v;
        a[i].u[v] = *(const unsigned int*)(base + kb);
      }
    }
    #pragma unroll
    for (int j = 0; j < 4; ++j) {
      const f16* base = &Bs[64 * wvn + 16 * j + l16][16 * half];
      #pragma unroll
      for (int v = 0; v < 8; ++v)         // B: lane=N, K=2v (+16 upper half)
        b[j].u[v] = *(const unsigned int*)(base + 2 * v);
    }
    #pragma unroll
    for (int i = 0; i < 2; ++i)
      #pragma unroll
      for (int j = 0; j < 4; ++j)
        acc[i][j] = __builtin_amdgcn_wmma_f32_16x16x32_f16(
            false, a[i].v, false, b[j].v, (short)0, acc[i][j], false, false);
    __syncthreads();
  }

  // epilogue: C layout lane=(N=l16, M=r+8*half); fast path for full tiles
  if (mBase + 128 <= Mrows) {
    #pragma unroll
    for (int j = 0; j < 4; ++j) {
      int n = nco + 64 * wvn + 16 * j + l16;
      float sc = (mat == 0) ? dsc[n % DH] : 1.0f;
      #pragma unroll
      for (int i = 0; i < 2; ++i)
        #pragma unroll
        for (int r = 0; r < 8; ++r) {
          int m = mBase + 32 * wvm + 16 * i + r + 8 * half;
          Out[(size_t)m * HID + n] = (f16)(acc[i][j][r] * sc);
        }
    }
  } else {
    #pragma unroll
    for (int j = 0; j < 4; ++j) {
      int n = nco + 64 * wvn + 16 * j + l16;
      float sc = (mat == 0) ? dsc[n % DH] : 1.0f;
      #pragma unroll
      for (int i = 0; i < 2; ++i)
        #pragma unroll
        for (int r = 0; r < 8; ++r) {
          int m = mBase + 32 * wvm + 16 * i + r + 8 * half;
          if (m < Mrows) Out[(size_t)m * HID + n] = (f16)(acc[i][j][r] * sc);
        }
    }
  }
}

// ---------------------------------------------------------------------------
// Attention: one wave32 per (b, h, u-block). k and sin_emb tiles staged via
// the Tensor Data Mover (async, TENSORcnt); v transposed manually.
// WMMA: q@k^T (2x6), q@sinT (6), probs@v (12).
// ---------------------------------------------------------------------------
__global__ __launch_bounds__(32) void gemma3n_attn_kernel(
    const f16* __restrict__ Q, const f16* __restrict__ Kd,
    const f16* __restrict__ Vd, const unsigned char* __restrict__ msk,
    const f16* __restrict__ semb, float* __restrict__ Out, int T) {
  __shared__ f16   kls[KROW * 32];  // k: [ctx][feat 192], TDM-padded stride 204
  __shared__ f16   sls[KROW * 16];  // sin_emb head slice (13 rows valid)
  __shared__ f16   vls[DH * 40];    // v transposed: [feat][ctx 32], stride 40
  __shared__ float acl[16 * 32];
  __shared__ float bdl[16 * 16];
  __shared__ f16   pls[16 * 34];

  const int lane = threadIdx.x;
  const int half = lane >> 4;
  const int l16  = lane & 15;
  const int u0 = blockIdx.x, h = blockIdx.y, bb = blockIdx.z;
  const int t0  = u0 * CHUNK;
  const int tc0 = t0 - PASTW;

  for (int i = lane; i < 16 * 34; i += 32) pls[i] = (f16)0.f;

  bool k_done = false, s_done = false;
#if USE_TDM
  if (tc0 >= 0 && tc0 + CTXW <= T) {      // interior block: pure 2-D tile
    tdm_load_2d_f16(Kd + ((size_t)(bb * T + tc0)) * HID + h * DH,
                    (unsigned)(size_t)&kls[0], CTXW, HID);
    k_done = true;
  }
  tdm_load_2d_f16(semb + h * DH, (unsigned)(size_t)&sls[0], 13, HID);
  s_done = true;
#endif
  if (!k_done) {                           // edge blocks: clamped manual copy
    for (int i = lane; i < CTXW * 48; i += 32) {
      int c = i / 48, d = (i % 48) * 2;
      int t = tc0 + c; t = t < 0 ? 0 : (t >= T ? T - 1 : t);
      unsigned off = KROW * c + d + 2 * (d >> 5);
      *(unsigned int*)&kls[off] =
          *(const unsigned int*)(Kd + ((size_t)(bb * T + t)) * HID + h * DH + d);
    }
  }
  if (!s_done) {
    for (int i = lane; i < 13 * 48; i += 32) {
      int f = i / 48, d = (i % 48) * 2;
      unsigned off = KROW * f + d + 2 * (d >> 5);
      *(unsigned int*)&sls[off] =
          *(const unsigned int*)(semb + (size_t)f * HID + h * DH + d);
    }
  }
  // v transposed (needed column-major for probs@v B-fragments)
  for (int i = lane; i < CTXW * 48; i += 32) {
    int c = i / 48, d = (i % 48) * 2;
    int t = tc0 + c; t = t < 0 ? 0 : (t >= T ? T - 1 : t);
    U32H uv; uv.w = *(const unsigned int*)(Vd + ((size_t)(bb * T + t)) * HID + h * DH + d);
    vls[(d + 0) * 40 + c] = uv.h[0];
    vls[(d + 1) * 40 + c] = uv.h[1];
  }

  // q A-fragments straight from global (pairs contiguous in feature dim)
  FragH qf[6];
  {
    int m = l16, tq = t0 + m;
    bool qok = (m < CHUNK) && (tq < T);
    const f16* qrow = Q + ((size_t)(bb * T + (qok ? tq : 0))) * HID + h * DH;
    #pragma unroll
    for (int s = 0; s < 6; ++s)
      #pragma unroll
      for (int v = 0; v < 8; ++v) {
        int kb = 32 * s + 16 * half + 2 * v;
        qf[s].u[v] = qok ? *(const unsigned int*)(qrow + kb) : 0u;
      }
  }
#if USE_TDM
  __builtin_amdgcn_s_wait_tensorcnt(0);    // TDM tiles landed in LDS
#endif
  __syncthreads();

  v8f ac0 = {}, ac1 = {}, bd = {};
  #pragma unroll
  for (int s = 0; s < 6; ++s) {            // K=192 in 6 steps of 32
    FragH b0, b1, bs;
    #pragma unroll
    for (int v = 0; v < 8; ++v) {
      int kk = 32 * s + 16 * half + 2 * v;
      unsigned o0 = KROW * l16 + kk + 2 * s;          // padded-layout offset
      unsigned o1 = KROW * (16 + l16) + kk + 2 * s;
      b0.u[v] = *(const unsigned int*)&kls[o0];
      b1.u[v] = *(const unsigned int*)&kls[o1];
      bs.u[v] = *(const unsigned int*)&sls[o0];
    }
    ac0 = __builtin_amdgcn_wmma_f32_16x16x32_f16(false, qf[s].v, false, b0.v, (short)0, ac0, false, false);
    ac1 = __builtin_amdgcn_wmma_f32_16x16x32_f16(false, qf[s].v, false, b1.v, (short)0, ac1, false, false);
    bd  = __builtin_amdgcn_wmma_f32_16x16x32_f16(false, qf[s].v, false, bs.v, (short)0, bd,  false, false);
  }
  #pragma unroll
  for (int r = 0; r < 8; ++r) {
    int m = r + 8 * half;
    acl[m * 32 + l16]      = ac0[r];
    acl[m * 32 + 16 + l16] = ac1[r];
    bdl[m * 16 + l16]      = bd[r];
  }
  __syncthreads();

  // per-row rel-shift + tanh soft-cap + mask + softmax (lanes 0..11)
  if (lane < CHUNK && (t0 + lane) < T) {
    const int w = lane;
    float xs[CTXW], mx = -3.0e38f;
    #pragma unroll
    for (int c = 0; c < CTXW; ++c) {
      int f = c - w;                       // rel-shift: bd[w][c-w], valid 0..12
      int tctx = tc0 + c;
      bool ok = (f >= 0) && (f <= PASTW) && (tctx >= 0) && (tctx < T) &&
                (msk[(size_t)bb * T + tctx] == 0);
      float v = -3.0e38f;
      if (ok) {
        v = acl[w * 32 + c] + bdl[w * 16 + f];
        v = tanhf(v * (1.0f / 50.0f)) * 50.0f;
        if (v > mx) mx = v;
      }
      xs[c] = v;
    }
    float sum = 0.f;
    #pragma unroll
    for (int c = 0; c < CTXW; ++c) {
      float e = (xs[c] > -1e38f) ? __expf(xs[c] - mx) : 0.f;
      xs[c] = e; sum += e;
    }
    float inv = (sum > 0.f) ? 1.0f / sum : 0.f;
    #pragma unroll
    for (int c = 0; c < CTXW; ++c) pls[w * 34 + c] = (f16)(xs[c] * inv);
  }
  __syncthreads();

  // out = probs @ v : A = probs (16x32), B = v (32 x 192 in 12 N tiles)
  FragH pf;
  #pragma unroll
  for (int v = 0; v < 8; ++v) {
    int kb = 16 * half + 2 * v;
    pf.u[v] = *(const unsigned int*)&pls[l16 * 34 + kb];
  }
  #pragma unroll
  for (int j = 0; j < 12; ++j) {
    FragH bv;
    #pragma unroll
    for (int v = 0; v < 8; ++v) {
      int c = 16 * half + 2 * v;
      bv.u[v] = *(const unsigned int*)&vls[(16 * j + l16) * 40 + c];
    }
    v8f o = {};
    o = __builtin_amdgcn_wmma_f32_16x16x32_f16(false, pf.v, false, bv.v, (short)0, o, false, false);
    #pragma unroll
    for (int r = 0; r < 8; ++r) {
      int m = r + 8 * half, tq = t0 + m;
      if (m < CHUNK && tq < T)
        Out[((size_t)(bb * T + tq)) * HID + h * DH + 16 * j + l16] = o[r];
    }
  }
}

// ---------------------------------------------------------------------------
extern "C" void kernel_launch(void* const* d_in, const int* in_sizes, int n_in,
                              void* d_out, int out_size, void* d_ws, size_t ws_size,
                              hipStream_t stream) {
  const float*         hs   = (const float*)d_in[0];
  const unsigned char* mask = (const unsigned char*)d_in[1];
  const float*         qw   = (const float*)d_in[2];
  const float*         kw   = (const float*)d_in[3];
  const float*         vw   = (const float*)d_in[4];
  const float*         pw   = (const float*)d_in[5];
  const float*         pds  = (const float*)d_in[6];
  float*               out  = (float*)d_out;

  const int Bb = 8;                         // batch (reference fixed)
  const int BT = in_sizes[1];               // b * t
  const int T  = BT / Bb;
  const int U  = (T + CHUNK - 1) / CHUNK;

  char* ws = (char*)d_ws;
  size_t qkvBytes = (size_t)BT * HID * sizeof(f16);
  f16*   Q    = (f16*)(ws);
  f16*   Kd   = (f16*)(ws + qkvBytes);
  f16*   Vd   = (f16*)(ws + 2 * qkvBytes);
  f16*   semb = (f16*)(ws + 3 * qkvBytes);
  float* dsc  = (float*)(ws + 3 * qkvBytes + 65536);

  gemma3n_dscale_kernel<<<1, 192, 0, stream>>>(pds, dsc);
  gemma3n_posemb_kernel<<<13, 256, 0, stream>>>(pw, semb);
  gemma3n_qkv_gemm_kernel<<<dim3(36, (BT + 127) / 128), 256, 0, stream>>>(
      hs, qw, kw, vw, dsc, Q, Kd, Vd, BT);
  gemma3n_attn_kernel<<<dim3(U, HEADS, Bb), 32, 0, stream>>>(
      Q, Kd, Vd, mask, semb, out, T);
}
